// SALA_64295660421358
// MI455X (gfx1250) — compile-verified, matrix operand order checked by reading, not run on hardware
//
#include <hip/hip_runtime.h>

typedef __attribute__((ext_vector_type(16))) _Float16 v16h;
typedef __attribute__((ext_vector_type(8)))  float    v8f;

#define C_DIM 256
#define HW    50176
#define W_DIM 224
#define NHEAD 8
#define HDIM  32
#define NWIN  1024   // 32x32 windows of 7x7
#define NTOK  49

__device__ __forceinline__ unsigned pack_f16(float a, float b) {
    union { _Float16 h[2]; unsigned u; } r;
    r.h[0] = (_Float16)a; r.h[1] = (_Float16)b;
    return r.u;
}

// K-offset of VGPR v for the 16-bit A/B fragment layout (ISA 7.12.2):
// VGPR 0..3 hold K = g*8 + {0,2,4,6}(+1), VGPR 4..7 hold K = 16 + g*8 + ...
__device__ __forceinline__ int k0_of(int v, int g) {
    return (v < 4) ? (g * 8 + 2 * v) : (16 + g * 8 + 2 * (v - 4));
}

__device__ __forceinline__ v8f wmma16(const unsigned* au, const unsigned* bu, v8f c) {
    union { unsigned u[8]; v16h h; } A, B;
#pragma unroll
    for (int i = 0; i < 8; i++) { A.u[i] = au[i]; B.u[i] = bu[i]; }
    return __builtin_amdgcn_wmma_f32_16x16x32_f16(false, A.h, false, B.h,
                                                  (short)0, c, false, false);
}

// ---------------- Kernel 0: convert the four 256x256 f32 weight matrices to
// f16, channel pairs packed into u32 so A fragments load with one b32 each.
__global__ void k_convert_w(const float* Wq, const float* Wk, const float* Wv,
                            const float* Wo, unsigned* wbf) {
    const int total = 4 * 256 * 128;
    for (int i = blockIdx.x * blockDim.x + threadIdx.x; i < total;
         i += gridDim.x * blockDim.x) {
        int m = i >> 15;
        int rem = i & 32767;
        int o = rem >> 7, c2 = rem & 127;
        const float* w = (m == 0) ? Wq : (m == 1) ? Wk : (m == 2) ? Wv : Wo;
        wbf[i] = pack_f16(w[o * 256 + 2 * c2], w[o * 256 + 2 * c2 + 1]);
    }
}

// ---------------- Kernel 1: QKV projection (GEMM over channels) fused with
// windowed-layout scatter. One block = 16 pixels, 4 waves share LDS x tile.
__global__ __launch_bounds__(128) void k_qkv(const float* __restrict__ x,
                                             const unsigned* __restrict__ wbf,
                                             _Float16* __restrict__ qw,
                                             _Float16* __restrict__ kw,
                                             _Float16* __restrict__ vw) {
    __shared__ unsigned xs[128 * 16];   // packed f16 pairs: [c2][pixel]
    const int b = blockIdx.y;
    const int nbase = blockIdx.x * 16;
    const int tid = threadIdx.x;
    for (int e = tid; e < 128 * 16; e += 128) {
        int c2 = e >> 4, p = e & 15;
        const float* xb = x + ((size_t)(b * C_DIM + 2 * c2)) * HW + nbase + p;
        xs[c2 * 16 + p] = pack_f16(xb[0], xb[HW]);
    }
    __syncthreads();
    const int wv = tid >> 5, lane = tid & 31, g = lane >> 4, ln = lane & 15;
    for (int j = wv; j < 48; j += 4) {          // 3 matrices x 16 M-tiles
        const int mat = j >> 4;
        const int mbase = (j & 15) * 16;
        v8f acc = {};
        const unsigned* wrow = wbf + mat * 32768 + (mbase + ln) * 128;
#pragma unroll
        for (int kb = 0; kb < 256; kb += 32) {
            unsigned au[8], bu[8];
#pragma unroll
            for (int v = 0; v < 8; v++) {
                int k0 = kb + k0_of(v, g);
                au[v] = wrow[k0 >> 1];
                bu[v] = xs[(k0 >> 1) * 16 + ln];
            }
            acc = wmma16(au, bu, acc);
        }
        _Float16* dst = (mat == 0) ? qw : (mat == 1) ? kw : vw;
        const int p = nbase + ln;
        const int h = p / W_DIM, w = p % W_DIM;
        const int win = (h / 7) * 32 + (w / 7);
        const int tok = (h % 7) * 7 + (w % 7);
#pragma unroll
        for (int r = 0; r < 8; r++) {
            int o = mbase + r + 8 * g;
            int head = o >> 5, d = o & 31;
            size_t addr = ((((size_t)b * NHEAD + head) * NWIN + win) * NTOK + tok) * HDIM + d;
            dst[addr] = (_Float16)acc[r];
        }
    }
}

// ---------------- Kernel 2: per-window attention. Block = (win, head, b).
// Tokens padded 49 -> 64 with zeros in LDS; scores masked before softmax.
__global__ __launch_bounds__(128) void k_attn(const _Float16* __restrict__ qw,
                                              const _Float16* __restrict__ kw,
                                              const _Float16* __restrict__ vw,
                                              _Float16* __restrict__ ao) {
    __shared__ _Float16 qs[64 * 32];         // [tok][d]       (A of S = Q K^T)
    __shared__ _Float16 ksh[16 * 64 * 2];    // [d2][tok][2]   (B of S)
    __shared__ _Float16 vsh[32 * 32 * 2];    // [t2][d][2]     (B of O = P V)
    __shared__ float    sc[64 * 64];         // f32 scores
    __shared__ _Float16 ph[64 * 64];         // f16 probabilities (A of O)
    const int win = blockIdx.x, head = blockIdx.y, b = blockIdx.z;
    const int tid = threadIdx.x;
    const size_t base = (((size_t)b * NHEAD + head) * NWIN + win) * (size_t)(NTOK * HDIM);
    for (int e = tid; e < 64 * 32; e += 128) {
        int tok = e >> 5, d = e & 31;
        _Float16 qv = (_Float16)0.f, kv = (_Float16)0.f, vv = (_Float16)0.f;
        if (tok < NTOK) {
            qv = qw[base + tok * HDIM + d];
            kv = kw[base + tok * HDIM + d];
            vv = vw[base + tok * HDIM + d];
        }
        qs[tok * 32 + d] = qv;
        ksh[((d >> 1) * 64 + tok) * 2 + (d & 1)] = kv;
        vsh[((tok >> 1) * 32 + d) * 2 + (tok & 1)] = vv;
    }
    __syncthreads();
    const int wv = tid >> 5, lane = tid & 31, g = lane >> 4, ln = lane & 15;
    const int mbase = wv * 16;
    const unsigned* qu = (const unsigned*)qs;
    const unsigned* ku = (const unsigned*)ksh;
    const unsigned* vu = (const unsigned*)vsh;
    const unsigned* pu = (const unsigned*)ph;
    const float scale = 0.17677669529663687f;   // 1/sqrt(32)
#pragma unroll
    for (int nt = 0; nt < 4; nt++) {            // S: K=32 = one WMMA
        unsigned au[8], bu[8];
#pragma unroll
        for (int v = 0; v < 8; v++) {
            int k0 = k0_of(v, g);
            au[v] = qu[(mbase + ln) * 16 + (k0 >> 1)];
            bu[v] = ku[(k0 >> 1) * 64 + nt * 16 + ln];
        }
        v8f z = {};
        v8f acc = wmma16(au, bu, z);
#pragma unroll
        for (int r = 0; r < 8; r++)
            sc[(mbase + r + 8 * g) * 64 + nt * 16 + ln] = acc[r] * scale;
    }
    __syncthreads();
    if (tid < 64) {                             // row-wise masked softmax
        int row = tid;
        float m = -1e30f;
        for (int c = 0; c < NTOK; c++) m = fmaxf(m, sc[row * 64 + c]);
        float s = 0.f;
        for (int c = 0; c < NTOK; c++) s += __expf(sc[row * 64 + c] - m);
        float inv = 1.0f / s;
        for (int c = 0; c < 64; c++)
            ph[row * 64 + c] = (c < NTOK) ? (_Float16)(__expf(sc[row * 64 + c] - m) * inv)
                                          : (_Float16)0.f;
    }
    __syncthreads();
    const int wh = win >> 5, ww = win & 31;
#pragma unroll
    for (int nt = 0; nt < 2; nt++) {            // O = P @ V : K = 64 = 2 WMMAs
        v8f acc = {};
#pragma unroll
        for (int ks = 0; ks < 2; ks++) {
            unsigned au[8], bu[8];
#pragma unroll
            for (int v = 0; v < 8; v++) {
                int k0 = ks * 32 + k0_of(v, g);
                au[v] = pu[(mbase + ln) * 32 + (k0 >> 1)];
                bu[v] = vu[(k0 >> 1) * 32 + nt * 16 + ln];
            }
            acc = wmma16(au, bu, acc);
        }
#pragma unroll
        for (int r = 0; r < 8; r++) {
            int qtok = mbase + r + 8 * g;
            if (qtok < NTOK) {
                int d = nt * 16 + ln;
                int h = wh * 7 + qtok / 7, w2 = ww * 7 + qtok % 7;
                int cch = head * HDIM + d;
                ao[((size_t)(b * C_DIM + cch)) * HW + h * W_DIM + w2] = (_Float16)acc[r];
            }
        }
    }
}

// ---------------- Kernel 3: Wo projection + residual -> y (f32)
__global__ __launch_bounds__(128) void k_wo(const _Float16* __restrict__ ao,
                                            const unsigned* __restrict__ wbf,
                                            const float* __restrict__ x,
                                            float* __restrict__ y) {
    __shared__ unsigned xs[128 * 16];
    const int b = blockIdx.y;
    const int nbase = blockIdx.x * 16;
    const int tid = threadIdx.x;
    for (int e = tid; e < 128 * 16; e += 128) {
        int c2 = e >> 4, p = e & 15;
        const _Float16* ab = ao + ((size_t)(b * C_DIM + 2 * c2)) * HW + nbase + p;
        union { _Float16 h[2]; unsigned u; } r;
        r.h[0] = ab[0]; r.h[1] = ab[HW];
        xs[c2 * 16 + p] = r.u;
    }
    __syncthreads();
    const int wv = tid >> 5, lane = tid & 31, g = lane >> 4, ln = lane & 15;
    for (int mt = wv; mt < 16; mt += 4) {
        const int mbase = mt * 16;
        v8f acc = {};
        const unsigned* wrow = wbf + 3 * 32768 + (mbase + ln) * 128;
#pragma unroll
        for (int kb = 0; kb < 256; kb += 32) {
            unsigned au[8], bu[8];
#pragma unroll
            for (int v = 0; v < 8; v++) {
                int k0 = kb + k0_of(v, g);
                au[v] = wrow[k0 >> 1];
                bu[v] = xs[(k0 >> 1) * 16 + ln];
            }
            acc = wmma16(au, bu, acc);
        }
        const int p = nbase + ln;
#pragma unroll
        for (int r = 0; r < 8; r++) {
            int o = mbase + r + 8 * g;
            size_t idx = ((size_t)(b * C_DIM + o)) * HW + p;
            y[idx] = acc[r] + x[idx];
        }
    }
}

// ---------------- Kernel 4: deterministic per-channel sum / sumsq
__global__ __launch_bounds__(256) void k_stats(const float* __restrict__ y,
                                               float* __restrict__ sums) {
    __shared__ float ls[256], lq[256];
    const int c = blockIdx.x;
    const int tid = threadIdx.x;
    float s = 0.f, q = 0.f;
    for (int b = 0; b < 4; b++) {
        const float* yb = y + ((size_t)(b * C_DIM + c)) * HW;
        for (int i = tid; i < HW; i += 256) {
            float v = yb[i];
            s += v; q += v * v;
        }
    }
    ls[tid] = s; lq[tid] = q;
    __syncthreads();
    for (int off = 128; off > 0; off >>= 1) {
        if (tid < off) { ls[tid] += ls[tid + off]; lq[tid] += lq[tid + off]; }
        __syncthreads();
    }
    if (tid == 0) { sums[c] = ls[0]; sums[256 + c] = lq[0]; }
}

// ---------------- Kernel 5: batchnorm normalize -> d_out
__global__ void k_bn(const float* __restrict__ y, const float* __restrict__ sums,
                     const float* __restrict__ gamma, const float* __restrict__ beta,
                     float* __restrict__ out, size_t n) {
    size_t i = (size_t)blockIdx.x * blockDim.x + threadIdx.x;
    if (i >= n) return;
    int c = (int)((i / HW) & (C_DIM - 1));
    const float invN = 1.0f / (4.0f * HW);
    float mean = sums[c] * invN;
    float var  = sums[256 + c] * invN - mean * mean;
    float r = rsqrtf(var + 1e-5f);
    out[i] = (y[i] - mean) * r * gamma[c] + beta[c];
}

extern "C" void kernel_launch(void* const* d_in, const int* in_sizes, int n_in,
                              void* d_out, int out_size, void* d_ws, size_t ws_size,
                              hipStream_t stream) {
    const float* x     = (const float*)d_in[0];
    const float* Wq    = (const float*)d_in[1];
    const float* Wk    = (const float*)d_in[2];
    const float* Wv    = (const float*)d_in[3];
    const float* Wo    = (const float*)d_in[4];
    const float* gamma = (const float*)d_in[5];
    const float* beta  = (const float*)d_in[6];

    char* ws = (char*)d_ws;
    float*    sums = (float*)ws;                          // 512 f32 (sum, sumsq)
    unsigned* wbf  = (unsigned*)(ws + 4096);              // 4*256*128 u32 = 512 KB
    size_t off = 4096 + 4ull * 256 * 128 * 4;
    const size_t qkv_elems = 4ull * NHEAD * NWIN * NTOK * HDIM;
    _Float16* qw = (_Float16*)(ws + off); off += qkv_elems * 2;
    _Float16* kw = (_Float16*)(ws + off); off += qkv_elems * 2;
    _Float16* vw = (_Float16*)(ws + off); off += qkv_elems * 2;
    _Float16* ao = (_Float16*)(ws + off); off += 4ull * C_DIM * HW * 2;
    float*    y  = (float*)(ws + off);

    hipLaunchKernelGGL(k_convert_w, dim3(256), dim3(256), 0, stream, Wq, Wk, Wv, Wo, wbf);
    hipLaunchKernelGGL(k_qkv,  dim3(HW / 16, 4), dim3(128), 0, stream, x, wbf, qw, kw, vw);
    hipLaunchKernelGGL(k_attn, dim3(NWIN, NHEAD, 4), dim3(128), 0, stream, qw, kw, vw, ao);
    hipLaunchKernelGGL(k_wo,   dim3(HW / 16, 4), dim3(128), 0, stream, ao, wbf, x, y);
    hipLaunchKernelGGL(k_stats, dim3(C_DIM), dim3(256), 0, stream, y, sums);
    size_t n = 4ull * C_DIM * HW;
    hipLaunchKernelGGL(k_bn, dim3((unsigned)((n + 255) / 256)), dim3(256), 0, stream,
                       y, sums, gamma, beta, (float*)d_out, n);
}
